// LeNet_DeBut_70153995812928
// MI455X (gfx1250) — matrix operand order repaired and checked by reading, not compiled, not run on hardware
//
#include <hip/hip_runtime.h>
#include <hip/hip_bf16.h>

typedef __attribute__((ext_vector_type(8)))  _Float16 v8h;
typedef __attribute__((ext_vector_type(16))) _Float16 v16h;
typedef __attribute__((ext_vector_type(8)))  float    v8f;

static __device__ __forceinline__ v16h combine16(v8h lo, v8h hi) {
    return __builtin_shufflevector(lo, hi, 0,1,2,3,4,5,6,7,8,9,10,11,12,13,14,15);
}

// A fragment (16-bit A 16x32, ISA 7.12.2): lane<16 -> M=lane, K{0..7},{16..23};
// lane>=16 -> M=lane-16, K{8..15},{24..31}. row points at this lane's M row.
static __device__ __forceinline__ v16h ld_a_frag(const _Float16* row, int kbase, int lane_hi) {
    const int o = lane_hi ? 8 : 0;
    v8h lo = *(const v8h*)(row + kbase + o);
    v8h hi = *(const v8h*)(row + kbase + 16 + o);
    return combine16(lo, hi);
}

// B fragment (wave32 B KxN): lane<16 -> N=lane, K{0..7},{16..23};
// lane>=16 -> N=lane-16, K{8..15},{24..31}. row points at this lane's N row of W (k-major).
static __device__ __forceinline__ v16h ld_b_frag(const _Float16* row, int kbase, int lane_hi) {
    const int o = lane_hi ? 8 : 0;
    v8h lo = *(const v8h*)(row + kbase + o);
    v8h hi = *(const v8h*)(row + kbase + 16 + o);
    return combine16(lo, hi);
}

// ---------------------------------------------------------------------------
// Kernel 1: one-time weight repack to f16.
//  w1h: fc1  [128][416]  (K 400->416 zero pad)
//  w2h: fc2  [64][128]
//  w3h: fc3  [16][64]    (rows 10->16 zero pad)
//  w2c: conv2 [16][96]   (OIHW rows are already k-major; K 72->96 zero pad)
// ---------------------------------------------------------------------------
__global__ void pack_weights(const float* __restrict__ f1w,
                             const float* __restrict__ f2w,
                             const float* __restrict__ f3w,
                             const float* __restrict__ c2w,
                             _Float16* __restrict__ w1h,
                             _Float16* __restrict__ w2h,
                             _Float16* __restrict__ w3h,
                             _Float16* __restrict__ w2c) {
    const int N1 = 128 * 416, N2 = 64 * 128, N3 = 16 * 64, N4 = 16 * 96;
    const int stride = gridDim.x * blockDim.x;
    for (int i = blockIdx.x * blockDim.x + threadIdx.x; i < N1 + N2 + N3 + N4; i += stride) {
        if (i < N1) {
            int n = i / 416, k = i % 416;
            w1h[i] = (_Float16)(k < 400 ? f1w[n * 400 + k] : 0.0f);
        } else if (i < N1 + N2) {
            int j = i - N1;
            w2h[j] = (_Float16)f2w[j];
        } else if (i < N1 + N2 + N3) {
            int j = i - N1 - N2;
            int n = j / 64, k = j % 64;
            w3h[j] = (_Float16)(n < 10 ? f3w[n * 64 + k] : 0.0f);
        } else {
            int j = i - N1 - N2 - N3;
            int n = j / 96, k = j % 96;
            w2c[j] = (_Float16)(k < 72 ? c2w[n * 72 + k] : 0.0f);
        }
    }
}

// ---------------------------------------------------------------------------
// Kernel 2: per-sample fused conv1+ReLU+pool2 -> conv2 (implicit GEMM on
// WMMA) + ReLU -> pool2.  One workgroup (256 threads = 8 waves) per sample.
//  - conv1 (K=9, N=8) stays on VALU: too small for a 16x16x32 tile.
//  - conv2 as GEMM: A = im2col [121->128 pos][72->96 K] f16 in LDS,
//    B = w2c [16][96]; each wave does one 16-row M-tile, 3 K-step WMMAs.
// Emits flattened activations [B, 416] f16 (400 real + 16 zero pad).
// ---------------------------------------------------------------------------
__global__ void __launch_bounds__(256) conv_fused(
    const float* __restrict__ x,
    const float* __restrict__ c1w, const float* __restrict__ c1b,
    const float* __restrict__ c2b,
    const _Float16* __restrict__ w2c,
    _Float16* __restrict__ act2) {
    __shared__ __attribute__((aligned(16))) float    sx[28 * 28];
    __shared__ float sw1[72], sb1[8], sb2[16];
    __shared__ float sp1[8 * 13 * 13];                      // conv1 pooled [8][13][13]
    __shared__ __attribute__((aligned(16))) _Float16 sA[128 * 96];   // im2col
    __shared__ __attribute__((aligned(16))) _Float16 sW[16 * 96];    // conv2 weights
    __shared__ __attribute__((aligned(16))) _Float16 sC[16 * 128];   // conv2 out [oc][pos]

    const int b = blockIdx.x, tid = threadIdx.x;
    const int wave    = tid >> 5;
    const int lane    = tid & 31;
    const int lane_hi = lane >> 4;
    const int lm      = lane & 15;

    // stage input sample (float4 -> global_load_b128), conv weights, biases
    {
        const float4* xs = (const float4*)(x + (size_t)b * 784);
        for (int i = tid; i < 196; i += 256) ((float4*)sx)[i] = xs[i];
    }
    for (int i = tid; i < 72; i += 256)  sw1[i] = c1w[i];
    for (int i = tid; i < 16 * 96; i += 256) sW[i] = w2c[i];
    if (tid < 8)  sb1[tid] = c1b[tid];
    if (tid < 16) sb2[tid] = c2b[tid];
    __syncthreads();

    // ---- conv1 (1ch, 3x3 valid -> 26x26) + ReLU + 2x2 pool -> [8][13][13] ----
    for (int i = tid; i < 8 * 13 * 13; i += 256) {
        int c = i / 169, r = i % 169, py = r / 13, px = r % 13;
        float m = 0.0f;  // ReLU outputs >= 0
        for (int dy = 0; dy < 2; ++dy)
            for (int dx = 0; dx < 2; ++dx) {
                int oy = 2 * py + dy, ox = 2 * px + dx;
                float s = sb1[c];
#pragma unroll
                for (int ky = 0; ky < 3; ++ky)
#pragma unroll
                    for (int kx = 0; kx < 3; ++kx)
                        s += sw1[c * 9 + ky * 3 + kx] * sx[(oy + ky) * 28 + ox + kx];
                m = fmaxf(m, fmaxf(s, 0.0f));
            }
        sp1[i] = m;
    }
    __syncthreads();

    // ---- im2col for conv2: A[pos=oy*11+ox][k=ic*9+ky*3+kx], zero padded ----
    for (int i = tid; i < 128 * 96; i += 256) {
        int p = i / 96, k = i % 96;
        float v = 0.0f;
        if (p < 121 && k < 72) {
            int py = p / 11, px = p % 11;
            int ic = k / 9, kk = k % 9, ky = kk / 3, kx = kk % 3;
            v = sp1[ic * 169 + (py + ky) * 13 + (px + kx)];
        }
        sA[i] = (_Float16)v;
    }
    __syncthreads();

    // ---- conv2 GEMM: [128,96] x [96,16] on WMMA; one M-tile per wave ----
    {
        const _Float16* arow = &sA[(wave * 16 + lm) * 96];  // this lane's position row
        const _Float16* wrow = &sW[lm * 96];                // this lane's oc row
        v8f acc = {};
#pragma unroll
        for (int ks = 0; ks < 3; ++ks) {
            v16h a  = ld_a_frag(arow, ks * 32, lane_hi);
            v16h bf = ld_b_frag(wrow, ks * 32, lane_hi);
            acc = __builtin_amdgcn_wmma_f32_16x16x32_f16(false, a, false, bf,
                                                         (short)0, acc, false, false);
        }
        const int   oc   = lm;            // D cols = output channel
        const float bias = sb2[oc];
        // Unconditional store: positions 121..127 hold relu(bias) garbage from
        // zero-padded im2col rows; pooling only reads positions <= 108, and
        // sC is sized [16][128], so no guard (and no divergent epilogue) needed.
#pragma unroll
        for (int r = 0; r < 8; ++r) {
            int p = wave * 16 + (lane_hi ? (8 + r) : r);  // D rows = position
            sC[oc * 128 + p] = (_Float16)fmaxf(acc[r] + bias, 0.0f);
        }
    }
    __syncthreads();

    // ---- 2x2 maxpool (11x11 -> 5x5) + flatten [16*25] + zero pad to 416 ----
    for (int i = tid; i < 400; i += 256) {
        int oc = i / 25, r = i % 25, qy = r / 5, qx = r % 5;
        float m = 0.0f;  // inputs are post-ReLU, >= 0
        for (int dy = 0; dy < 2; ++dy)
            for (int dx = 0; dx < 2; ++dx)
                m = fmaxf(m, (float)sC[oc * 128 + (2 * qy + dy) * 11 + (2 * qx + dx)]);
        act2[(size_t)b * 416 + i] = (_Float16)m;
    }
    if (tid < 16) act2[(size_t)b * 416 + 400 + tid] = (_Float16)0.0f;
}

// ---------------------------------------------------------------------------
// Kernel 3: fused FC chain on WMMA. 8 waves/block; each wave owns a 16-row
// batch tile. fc1 (416->128, ReLU) -> h1 in LDS (f16) -> fc2 (128->64) ->
// h2 in LDS -> fc3 (64->16 padded, store cols 0..9).
// ---------------------------------------------------------------------------
__global__ void __launch_bounds__(256) fc_fused(
    const _Float16* __restrict__ act2,
    const _Float16* __restrict__ w1h, const float* __restrict__ b1,
    const _Float16* __restrict__ w2h, const float* __restrict__ b2,
    const _Float16* __restrict__ w3h, const float* __restrict__ b3,
    float* __restrict__ out) {
    __shared__ __attribute__((aligned(16))) _Float16 h1[8][16 * 128];
    __shared__ __attribute__((aligned(16))) _Float16 h2[8][16 * 64];

    const int wave    = threadIdx.x >> 5;
    const int lane    = threadIdx.x & 31;
    const int lane_hi = lane >> 4;
    const int lm      = lane & 15;
    const int row0    = (blockIdx.x * 8 + wave) * 16;

    const _Float16* arow = act2 + (size_t)(row0 + lm) * 416;  // this lane's M row

    // ---- fc1: [16,416] x [416,128] + bias, ReLU -> h1 ----
    for (int nt = 0; nt < 8; ++nt) {
        v8f acc = {};
        const _Float16* wrow = w1h + (size_t)(nt * 16 + lm) * 416;
#pragma unroll
        for (int ks = 0; ks < 13; ++ks) {
            v16h a  = ld_a_frag(arow, ks * 32, lane_hi);
            v16h bf = ld_b_frag(wrow, ks * 32, lane_hi);
            acc = __builtin_amdgcn_wmma_f32_16x16x32_f16(false, a, false, bf,
                                                         (short)0, acc, false, false);
        }
        const int   n    = nt * 16 + lm;
        const float bias = b1[n];
#pragma unroll
        for (int r = 0; r < 8; ++r) {
            int m = lane_hi ? (8 + r) : r;  // D layout: lanes>=16 hold M=8..15
            h1[wave][m * 128 + n] = (_Float16)fmaxf(acc[r] + bias, 0.0f);
        }
    }

    // ---- fc2: [16,128] x [128,64] + bias -> h2 (no ReLU) ----
    const _Float16* h1row = &h1[wave][lm * 128];
    for (int nt = 0; nt < 4; ++nt) {
        v8f acc = {};
        const _Float16* wrow = w2h + (size_t)(nt * 16 + lm) * 128;
#pragma unroll
        for (int ks = 0; ks < 4; ++ks) {
            v16h a  = ld_a_frag(h1row, ks * 32, lane_hi);
            v16h bf = ld_b_frag(wrow, ks * 32, lane_hi);
            acc = __builtin_amdgcn_wmma_f32_16x16x32_f16(false, a, false, bf,
                                                         (short)0, acc, false, false);
        }
        const int   n    = nt * 16 + lm;
        const float bias = b2[n];
#pragma unroll
        for (int r = 0; r < 8; ++r) {
            int m = lane_hi ? (8 + r) : r;
            h2[wave][m * 64 + n] = (_Float16)(acc[r] + bias);
        }
    }

    // ---- fc3: [16,64] x [64,16pad] + bias -> out cols 0..9 ----
    {
        const _Float16* h2row = &h2[wave][lm * 64];
        const _Float16* wrow  = w3h + (size_t)lm * 64;
        v8f acc = {};
#pragma unroll
        for (int ks = 0; ks < 2; ++ks) {
            v16h a  = ld_a_frag(h2row, ks * 32, lane_hi);
            v16h bf = ld_b_frag(wrow, ks * 32, lane_hi);
            acc = __builtin_amdgcn_wmma_f32_16x16x32_f16(false, a, false, bf,
                                                         (short)0, acc, false, false);
        }
        const int n = lm;
        if (n < 10) {
            const float bias = b3[n];
#pragma unroll
            for (int r = 0; r < 8; ++r) {
                int m = lane_hi ? (8 + r) : r;
                out[(size_t)(row0 + m) * 10 + n] = acc[r] + bias;
            }
        }
    }
}

// ---------------------------------------------------------------------------
extern "C" void kernel_launch(void* const* d_in, const int* in_sizes, int n_in,
                              void* d_out, int out_size, void* d_ws, size_t ws_size,
                              hipStream_t stream) {
    const float* x   = (const float*)d_in[0];
    const float* c1w = (const float*)d_in[1];
    const float* c1b = (const float*)d_in[2];
    const float* c2w = (const float*)d_in[3];
    const float* c2b = (const float*)d_in[4];
    const float* f1w = (const float*)d_in[5];
    const float* f1b = (const float*)d_in[6];
    const float* f2w = (const float*)d_in[7];
    const float* f2b = (const float*)d_in[8];
    const float* f3w = (const float*)d_in[9];
    const float* f3b = (const float*)d_in[10];
    float* out = (float*)d_out;

    const int B = in_sizes[0] / 784;  // 16384

    char* ws = (char*)d_ws;
    _Float16* act2 = (_Float16*)ws;  ws += (size_t)B * 416 * sizeof(_Float16);
    _Float16* w1h  = (_Float16*)ws;  ws += (size_t)128 * 416 * sizeof(_Float16);
    _Float16* w2h  = (_Float16*)ws;  ws += (size_t)64 * 128 * sizeof(_Float16);
    _Float16* w3h  = (_Float16*)ws;  ws += (size_t)16 * 64 * sizeof(_Float16);
    _Float16* w2c  = (_Float16*)ws;

    pack_weights<<<64, 256, 0, stream>>>(f1w, f2w, f3w, c2w, w1h, w2h, w3h, w2c);
    conv_fused<<<B, 256, 0, stream>>>(x, c1w, c1b, c2b, w2c, act2);
    fc_fused<<<B / 128, 256, 0, stream>>>(act2, w1h, f1b, w2h, f2b, w3h, f3b, out);
}
